// AttentionPool_5248450035828
// MI455X (gfx1250) — matrix operand order repaired and checked by reading, not compiled, not run on hardware
//
#include <hip/hip_runtime.h>

typedef __attribute__((ext_vector_type(16))) __bf16        v16bf;
typedef __attribute__((ext_vector_type(8)))  float         v8f;
typedef __attribute__((ext_vector_type(8)))  unsigned int  v8u;

#define IN_DIM 128
#define HID     64

// ---------- helpers ----------

// exact RNE f32 -> bf16 bits (used once, in prep kernel)
__device__ __forceinline__ unsigned bf_bits(float f) {
    unsigned u = __float_as_uint(f);
    u += 0x7FFFu + ((u >> 16) & 1u);
    return u >> 16;
}

// fast packed f32x2 -> bf16x2: round-half-up + single v_perm_b32
__device__ __forceinline__ unsigned pk_bf(float lo, float hi) {
    unsigned a = __float_as_uint(hi) + 0x8000u;
    unsigned b = __float_as_uint(lo) + 0x8000u;
    return __builtin_amdgcn_perm(a, b, 0x07060302u);  // {a.hi16, b.hi16}
}

__device__ __forceinline__ void atomicMaxF(float* addr, float val) {
    if (val >= 0.0f) atomicMax((int*)addr, __float_as_int(val));
    else             atomicMin((unsigned int*)addr, __float_as_uint(val));
}

// ---------- kernel 0: init out / m / denom ----------

__global__ void init_kernel(float* __restrict__ out, float* __restrict__ m,
                            float* __restrict__ denom, int g, int total) {
    int i = blockIdx.x * blockDim.x + threadIdx.x;
    if (i < total) out[i] = 0.0f;
    if (i < g) { m[i] = -__builtin_inff(); denom[i] = 0.0f; }
}

// ---------- kernel 0b: pre-pack W1 into bf16 WMMA B-matrix lane layout ----------
// Bp[tile][lane][dword v]: tile = kt*4+nt; element e=2v+{0,1};
// K = 32*kt + (lane<16?0:16) + e ; N = 16*nt + (lane&15)

__global__ void prep_w1_kernel(const float* __restrict__ W1, unsigned* __restrict__ Bp) {
    int idx = blockIdx.x * blockDim.x + threadIdx.x;   // 0..4095
    int t    = idx >> 8;          // tile 0..15
    int kt   = t >> 2;
    int nt   = t & 3;
    int rem  = idx & 255;
    int lane = rem >> 3;
    int v    = rem & 7;
    int nl   = lane & 15;
    int kb   = (lane < 16) ? 0 : 16;
    int k0   = 32 * kt + kb + 2 * v;
    int ncol = 16 * nt + nl;
    unsigned lo = bf_bits(W1[(size_t)k0 * HID + ncol]);
    unsigned hi = bf_bits(W1[(size_t)(k0 + 1) * HID + ncol]);
    Bp[idx] = lo | (hi << 16);
}

// ---------- kernel 1: gate MLP via WMMA bf16 ----------
// 1 wave = 16 rows; block = 256 threads = 8 waves = 128 rows.
// B (W1) comes pre-packed from workspace: one 32-byte vector load per tile per lane.

__global__ void gate_mlp_kernel(const float* __restrict__ x,
                                const v16bf* __restrict__ Bp,
                                const float* __restrict__ b1,
                                const float* __restrict__ W2,
                                const float* __restrict__ b2,
                                float* __restrict__ gate, int n) {
    const int lane = threadIdx.x & 31;
    const int wave = threadIdx.x >> 5;
    const int row0 = blockIdx.x * 128 + wave * 16;
    if (row0 >= n) return;                    // wave-uniform

    const int nl   = lane & 15;
    const int hi   = lane >> 4;
    const int kb_a = hi ? 16 : 0;             // A-matrix K base per half-wave

    // C init = bias b1
    v8f c[4];
#pragma unroll
    for (int nt = 0; nt < 4; ++nt) {
        float bv = b1[16 * nt + nl];
#pragma unroll
        for (int r = 0; r < 8; ++r) c[nt][r] = bv;
    }

    const float* xrow = x + (size_t)(row0 + nl) * IN_DIM;

#pragma unroll
    for (int kt = 0; kt < 4; ++kt) {
        // A tile 16x32 bf16: lane<16 holds K={0..15}, lane>=16 holds {16..31}
        const float* xp = xrow + 32 * kt + kb_a;
        float4 f0 = *(const float4*)(xp + 0);
        float4 f1 = *(const float4*)(xp + 4);
        float4 f2 = *(const float4*)(xp + 8);
        float4 f3 = *(const float4*)(xp + 12);
        v8u au;
        au[0] = pk_bf(f0.x, f0.y); au[1] = pk_bf(f0.z, f0.w);
        au[2] = pk_bf(f1.x, f1.y); au[3] = pk_bf(f1.z, f1.w);
        au[4] = pk_bf(f2.x, f2.y); au[5] = pk_bf(f2.z, f2.w);
        au[6] = pk_bf(f3.x, f3.y); au[7] = pk_bf(f3.z, f3.w);
        v16bf a = __builtin_bit_cast(v16bf, au);

#pragma unroll
        for (int nt = 0; nt < 4; ++nt) {
            v16bf bm = Bp[(kt * 4 + nt) * 32 + lane];   // pre-packed, L2-hot
            c[nt] = __builtin_amdgcn_wmma_f32_16x16x32_bf16(
                false, a, false, bm, (short)0, c[nt], false, false);
        }
    }

    // second layer: relu + dot with W2, cross-lane reduce within 16-lane halves
    float s[8] = {0, 0, 0, 0, 0, 0, 0, 0};
#pragma unroll
    for (int nt = 0; nt < 4; ++nt) {
        float w2v = W2[16 * nt + nl];
#pragma unroll
        for (int r = 0; r < 8; ++r) {
            float h = c[nt][r];
            h = h > 0.0f ? h : 0.0f;
            s[r] += h * w2v;
        }
    }
    const float bias2 = b2[0];
#pragma unroll
    for (int r = 0; r < 8; ++r) {
        float v = s[r];
        v += __shfl_xor(v, 1, 32);
        v += __shfl_xor(v, 2, 32);
        v += __shfl_xor(v, 4, 32);
        v += __shfl_xor(v, 8, 32);            // masks <16 stay within half-wave
        if (nl == 0) gate[row0 + r + hi * 8] = v + bias2;
    }
}

// ---------- kernel 2: segment max (batch sorted) ----------

__global__ void seg_max_kernel(const float* __restrict__ gate,
                               const long long* __restrict__ batch,
                               float* __restrict__ m, int n) {
    int i = blockIdx.x * blockDim.x + threadIdx.x;
    int lane = threadIdx.x & 31;
    int ic = i < n ? i : (n - 1);
    long long g = batch[ic];
    float v = (i < n) ? gate[i] : -__builtin_inff();
    long long gf = __shfl(g, 0, 32);
    long long gl = __shfl(g, 31, 32);
    if (gf == gl) {                            // sorted => whole wave in one segment
        v = fmaxf(v, __shfl_xor(v, 1, 32));
        v = fmaxf(v, __shfl_xor(v, 2, 32));
        v = fmaxf(v, __shfl_xor(v, 4, 32));
        v = fmaxf(v, __shfl_xor(v, 8, 32));
        v = fmaxf(v, __shfl_xor(v, 16, 32));
        if (lane == 0) atomicMaxF(&m[gf], v);
    } else if (i < n) {
        atomicMaxF(&m[g], v);
    }
}

// ---------- kernel 3: e = exp(gate - m[g]); denom = segment_sum(e) ----------

__global__ void seg_expsum_kernel(const float* __restrict__ gate,
                                  const long long* __restrict__ batch,
                                  const float* __restrict__ m,
                                  float* __restrict__ e,
                                  float* __restrict__ denom, int n) {
    int i = blockIdx.x * blockDim.x + threadIdx.x;
    int lane = threadIdx.x & 31;
    int ic = i < n ? i : (n - 1);
    long long g = batch[ic];
    float ex = 0.0f;
    if (i < n) { ex = expf(gate[i] - m[g]); e[i] = ex; }
    long long gf = __shfl(g, 0, 32);
    long long gl = __shfl(g, 31, 32);
    if (gf == gl) {
        float v = ex;
        v += __shfl_xor(v, 1, 32);
        v += __shfl_xor(v, 2, 32);
        v += __shfl_xor(v, 4, 32);
        v += __shfl_xor(v, 8, 32);
        v += __shfl_xor(v, 16, 32);
        if (lane == 0) atomicAdd(&denom[gf], v);
    } else if (i < n) {
        atomicAdd(&denom[g], ex);
    }
}

// ---------- kernel 4: out[g,:] += (e/denom[g]) * x[i,:], register-accumulated ----------

__global__ void scatter_kernel(const float* __restrict__ x,
                               const long long* __restrict__ batch,
                               const float* __restrict__ e,
                               const float* __restrict__ denom,
                               float* __restrict__ out, int n) {
    int d = threadIdx.x;                       // column 0..127
    int start = blockIdx.x * 256;
    if (start >= n) return;
    int end = start + 256; if (end > n) end = n;

    long long gcur = batch[start];
    float acc = 0.0f;
    for (int i = start; i < end; ++i) {
        long long gi = batch[i];               // block-uniform broadcast load
        if (gi != gcur) {
            atomicAdd(&out[(size_t)gcur * IN_DIM + d], acc / denom[gcur]);
            acc = 0.0f;
            gcur = gi;
        }
        acc += e[i] * x[(size_t)i * IN_DIM + d];
    }
    atomicAdd(&out[(size_t)gcur * IN_DIM + d], acc / denom[gcur]);
}

// ---------- launcher ----------

extern "C" void kernel_launch(void* const* d_in, const int* in_sizes, int n_in,
                              void* d_out, int out_size, void* d_ws, size_t ws_size,
                              hipStream_t stream) {
    const float*     x     = (const float*)d_in[0];
    const long long* batch = (const long long*)d_in[1];
    const float*     W1    = (const float*)d_in[2];
    const float*     b1    = (const float*)d_in[3];
    const float*     W2    = (const float*)d_in[4];
    const float*     b2    = (const float*)d_in[5];
    float* out = (float*)d_out;

    const int n = in_sizes[0] / IN_DIM;        // 2,000,000
    const int g = out_size / IN_DIM;           // 8,192

    // workspace (floats): gate[n] | e[n] | m[g] | denom[g] | Bpacked[4096 dwords]
    float* gate  = (float*)d_ws;
    float* evec  = gate + n;
    float* mbuf  = evec + n;
    float* denom = mbuf + g;
    unsigned* Bp = (unsigned*)(denom + g);     // 32-byte aligned (2n+2g divisible by 8)

    const int initN = g * IN_DIM;
    init_kernel<<<(initN + 255) / 256, 256, 0, stream>>>(out, mbuf, denom, g, initN);

    prep_w1_kernel<<<16, 256, 0, stream>>>(W1, Bp);

    gate_mlp_kernel<<<(n + 127) / 128, 256, 0, stream>>>(
        x, (const v16bf*)Bp, b1, W2, b2, gate, n);

    seg_max_kernel<<<(n + 255) / 256, 256, 0, stream>>>(gate, batch, mbuf, n);

    seg_expsum_kernel<<<(n + 255) / 256, 256, 0, stream>>>(gate, batch, mbuf, evec, denom, n);

    scatter_kernel<<<(n + 255) / 256, 128, 0, stream>>>(x, batch, evec, denom, out, n);
}
